// LSTM_36575941492857
// MI455X (gfx1250) — matrix operand order repaired and checked by reading, not compiled, not run on hardware
//
#include <hip/hip_runtime.h>
#include <hip/hip_bf16.h>
#include <math.h>
#include <stdint.h>

// ---------------------------------------------------------------------------
// LSTM on gfx1250 (MI455X), wave32 + v_wmma_f32_16x16x32_bf16.
//   Phase 1: convert W -> bf16 transposed Wt[n][k]; convert x -> bf16.
//   Phase 2: P[t,b,n] = x_t @ W_x   (big parallel WMMA GEMM, K=1024)
//   Phase 3: 512 sequential step kernels: gates = P[t] + h @ W_h + bias,
//            fused sigmoid/tanh/cell update in the WMMA C/D register layout.
// This revision:
//   * Tensor Data Mover (tensor_load_to_lds, TENSORcnt) stages both the
//     weight slab and the h state into LDS (iterate mode adds the bank-
//     conflict padding), with a manual global->LDS fallback when the builtin
//     is unavailable.
//   * Register double-buffering of BOTH A and B fragments so WMMAs overlap
//     the previous iteration's loads (no s_wait_*cnt 0 before wmma).
// ---------------------------------------------------------------------------

typedef __attribute__((ext_vector_type(16))) __bf16 v16bf;
typedef __attribute__((ext_vector_type(8)))  __bf16 v8bf;
typedef __attribute__((ext_vector_type(4)))  __bf16 v4bf;
typedef __attribute__((ext_vector_type(8)))  float  v8f;
typedef __attribute__((ext_vector_type(4)))  unsigned int u32x4;
typedef __attribute__((ext_vector_type(8)))  int          i32x8;
typedef __attribute__((ext_vector_type(4)))  int          i32x4;

#define T_STEPS 512
#define BATCH   64
#define IN_DIM  1024
#define HID     1024
#define G4      4096   // 4*HID
#define KTOT    2048   // IN_DIM + HID (rows of weight)

// LDS slab: rows of 1024 halves padded to stride 1032 halves (2064 B =
// 516 dwords; 516 % 64 = 4 -> 16 consecutive rows hit 16 distinct banks,
// and every 16B ds_load chunk stays 16B-aligned).
#define LDSB_STRIDE 1032

// ---- TDM availability ------------------------------------------------------
#ifndef CDNA5_USE_TDM
#if defined(__has_builtin)
#if __has_builtin(__builtin_amdgcn_tensor_load_to_lds) && \
    __has_builtin(__builtin_amdgcn_s_wait_tensorcnt)
#define CDNA5_USE_TDM 1
#endif
#endif
#endif
#ifndef CDNA5_USE_TDM
#define CDNA5_USE_TDM 0
#endif

#if CDNA5_USE_TDM
// One TDM descriptor: copy `nrows` rows of 1024 bf16 elements from global
// (row stride `gstride_elems`) into LDS at byte offset `lds_base`, using
// TDM iterate mode so each LDS row lands at stride LDSB_STRIDE elements.
__device__ __forceinline__ void tdm_load_rows(unsigned lds_base,
                                              const __bf16* gbase,
                                              int nrows, int gstride_elems) {
    unsigned long long ga = (unsigned long long)(uintptr_t)gbase;
    u32x4 g0 = {};
    g0[0] = 1u;                                           // count = 1 (valid)
    g0[1] = lds_base;                                     // lds_addr (bytes)
    g0[2] = (unsigned)ga;                                 // global_addr lo
    g0[3] = (unsigned)((ga >> 32) & 0x01ffffffu) | (2u << 30);  // hi | type=2
    i32x8 g1 = {};
    g1[0] = (1 << 16) | (1 << 19);        // data_size=2B, iterate_enable=1
    g1[1] = (int)(1024u << 16);           // tensor_dim0 = 1024 (elements)
    g1[2] = (int)(1u << 16);              // tensor_dim1 = 1
    g1[3] = (int)(1024u << 16);           // tile_dim0 = 1024
    g1[4] = 1;                            // tile_dim1 = 1
    g1[5] = gstride_elems;                // tensor_dim0_stride
    i32x4 g2 = {};
    g2[1] = LDSB_STRIDE;                  // lds_addr_increment (elements)
    g2[2] = gstride_elems;                // global_addr_increment (elements)
    g2[3] = (nrows - 1) << 16;            // iterate_count (0 => 1x)
    i32x4 g3 = {};
#if __clang_major__ >= 23
    i32x8 g4 = {};
    __builtin_amdgcn_tensor_load_to_lds(g0, g1, g2, g3, g4, 0);
#else
    __builtin_amdgcn_tensor_load_to_lds(g0, g1, g2, g3, 0);
#endif
}
#endif

// Manual fallback: same LDS layout, plain global_load + ds_store.
__device__ __forceinline__ void stage_rows_manual(__bf16* lds, const __bf16* gbase,
                                                  int nrows, int gstride_elems) {
    for (int idx = threadIdx.x; idx < nrows * 128; idx += blockDim.x) {
        int r  = idx >> 7;            // row
        int ch = idx & 127;           // 16B chunk within row
        *((v8bf*)(lds + (size_t)r * LDSB_STRIDE) + ch) =
            *((const v8bf*)(gbase + (size_t)r * gstride_elems) + ch);
    }
}

// ---- WMMA fragment helpers (per CDNA5 ISA 7.12.2 layouts, wave32) ----------

// A-matrix 16x32 bf16 (MxK) from global, row-major, leading dim `ld`.
__device__ __forceinline__ v16bf load_a_frag(const __bf16* tile, int ld) {
    int lane = threadIdx.x & 31;
    int m    = lane & 15;
    int kb   = (lane < 16) ? 0 : 8;
    const __bf16* p = tile + (size_t)m * ld + kb;
    v8bf lo = *(const v8bf*)(p);
    v8bf hi = *(const v8bf*)(p + 16);
    return __builtin_shufflevector(lo, hi,
        0,1,2,3,4,5,6,7,8,9,10,11,12,13,14,15);
}

// A fragment from the LDS slab (row0 = first slab row of this M-tile).
__device__ __forceinline__ v16bf load_a_lds(const __bf16* lds, int row0, int k0) {
    int lane = threadIdx.x & 31;
    int m    = lane & 15;
    int kb   = (lane < 16) ? 0 : 8;
    const __bf16* p = lds + (size_t)(row0 + m) * LDSB_STRIDE + k0 + kb;
    v8bf lo = *(const v8bf*)(p);
    v8bf hi = *(const v8bf*)(p + 16);
    return __builtin_shufflevector(lo, hi,
        0,1,2,3,4,5,6,7,8,9,10,11,12,13,14,15);
}

// B-matrix 32x16 bf16 (KxN) fragment from the LDS slab (transposed weight
// rows). lanes 0-15: N=lane, K=0..7,16..23 ; lanes 16-31: N=lane-16, K=8..15,24..31.
__device__ __forceinline__ v16bf load_b_lds(const __bf16* lds, int row0, int k0) {
    int lane = threadIdx.x & 31;
    int n    = lane & 15;
    int kb   = (lane < 16) ? 0 : 8;
    const __bf16* p = lds + (size_t)(row0 + n) * LDSB_STRIDE + k0 + kb;
    v8bf lo = *(const v8bf*)(p);
    v8bf hi = *(const v8bf*)(p + 16);
    return __builtin_shufflevector(lo, hi,
        0,1,2,3,4,5,6,7,8,9,10,11,12,13,14,15);
}

__device__ __forceinline__ v8f wmma_bf16(v16bf a, v16bf b, v8f c) {
    return __builtin_amdgcn_wmma_f32_16x16x32_bf16(
        false, a, false, b, (short)0, c, false, false);
}

__device__ __forceinline__ float sigmoidf_(float x) {
    return 1.0f / (1.0f + __expf(-x));
}

// ---- Phase 1 kernels -------------------------------------------------------

__global__ void k_wt_convert(const float* __restrict__ W, __bf16* __restrict__ Wt) {
    size_t i = (size_t)blockIdx.x * blockDim.x + threadIdx.x;  // over KTOT*G4
    int k = (int)(i >> 12);
    int n = (int)(i & 4095);
    Wt[(size_t)n * KTOT + k] = (__bf16)W[i];
}

__global__ void k_x_convert(const float* __restrict__ x, __bf16* __restrict__ xb) {
    size_t i = (size_t)blockIdx.x * blockDim.x + threadIdx.x;
    const float4 v = ((const float4*)x)[i];
    v4bf o;
    o[0] = (__bf16)v.x; o[1] = (__bf16)v.y; o[2] = (__bf16)v.z; o[3] = (__bf16)v.w;
    ((v4bf*)xb)[i] = o;
}

__global__ void k_init(const float* __restrict__ h0, const float* __restrict__ c0,
                       __bf16* __restrict__ hb, float* __restrict__ hf,
                       float* __restrict__ cb) {
    int i = blockIdx.x * blockDim.x + threadIdx.x;
    int j = i & (HID - 1);
    float h = h0[j];
    hb[i] = (__bf16)h;
    hf[i] = h;
    cb[i] = c0[j];
}

// ---- Phase 2: x-projection GEMM  P = xb @ Wx  (bf16 out) -------------------
__global__ void __launch_bounds__(256) k_xproj(const __bf16* __restrict__ xb,
                                               const __bf16* __restrict__ Wt,
                                               __bf16* __restrict__ P) {
    __shared__ __bf16 ldsB[64 * LDSB_STRIDE];

    int n0    = (blockIdx.x & 63) * 64;        // N-group (64 cols)
    int mtile = (blockIdx.x >> 6) * 8 + (threadIdx.x >> 5);
    int lane  = threadIdx.x & 31;

#if CDNA5_USE_TDM
    if ((threadIdx.x >> 5) == 0) {
        tdm_load_rows((unsigned)(uintptr_t)ldsB,
                      Wt + (size_t)n0 * KTOT, 64, KTOT);
        __builtin_amdgcn_s_wait_tensorcnt(0);
    }
#else
    stage_rows_manual(ldsB, Wt + (size_t)n0 * KTOT, 64, KTOT);
#endif
    __syncthreads();

    v8f acc[4] = {};
    const __bf16* arow = xb + (size_t)(mtile * 16) * IN_DIM;

    v16bf a_cur = load_a_frag(arow, IN_DIM);
    v16bf b0 = load_b_lds(ldsB,  0, 0);
    v16bf b1 = load_b_lds(ldsB, 16, 0);
    v16bf b2 = load_b_lds(ldsB, 32, 0);
    v16bf b3 = load_b_lds(ldsB, 48, 0);
    int k0 = 0;
    for (; k0 < IN_DIM - 32; k0 += 32) {
        v16bf a_n = load_a_frag(arow + k0 + 32, IN_DIM);
        v16bf c0 = load_b_lds(ldsB,  0, k0 + 32);
        v16bf c1 = load_b_lds(ldsB, 16, k0 + 32);
        v16bf c2 = load_b_lds(ldsB, 32, k0 + 32);
        v16bf c3 = load_b_lds(ldsB, 48, k0 + 32);
        acc[0] = wmma_bf16(a_cur, b0, acc[0]);
        acc[1] = wmma_bf16(a_cur, b1, acc[1]);
        acc[2] = wmma_bf16(a_cur, b2, acc[2]);
        acc[3] = wmma_bf16(a_cur, b3, acc[3]);
        a_cur = a_n; b0 = c0; b1 = c1; b2 = c2; b3 = c3;
    }
    acc[0] = wmma_bf16(a_cur, b0, acc[0]);
    acc[1] = wmma_bf16(a_cur, b1, acc[1]);
    acc[2] = wmma_bf16(a_cur, b2, acc[2]);
    acc[3] = wmma_bf16(a_cur, b3, acc[3]);

    int n  = lane & 15;
    int mb = (lane >> 4) << 3;
#pragma unroll
    for (int j = 0; j < 4; ++j) {
#pragma unroll
        for (int r = 0; r < 8; ++r) {
            int m = mb + r;
            P[(size_t)(mtile * 16 + m) * G4 + (n0 + j * 16 + n)] = (__bf16)acc[j][r];
        }
    }
}

// ---- Phase 3: one recurrent step -------------------------------------------
// grid = 64 blocks (hidden-col tiles of 16), block = 128 threads = 4 waves.
__global__ void __launch_bounds__(128) k_step(
    const __bf16* __restrict__ hin,
    __bf16*       __restrict__ hout,
    float*        __restrict__ hfp,
    float*        __restrict__ cbuf,
    const __bf16* __restrict__ Pt,
    const __bf16* __restrict__ Wt,
    const float*  __restrict__ bias,
    float*        __restrict__ out_t)
{
    __shared__ __bf16 ldsA[64 * LDSB_STRIDE];   // h state   (64 x 1024)
    __shared__ __bf16 ldsB[64 * LDSB_STRIDE];   // weight slab (4 gates x 16)

    int w    = threadIdx.x >> 5;      // 0..3 -> batch tile
    int lane = threadIdx.x & 31;
    int j0   = blockIdx.x * 16;       // hidden column tile

#if CDNA5_USE_TDM
    if (w == 0) {
        tdm_load_rows((unsigned)(uintptr_t)ldsA, hin, 64, HID);
#pragma unroll
        for (int g = 0; g < 4; ++g)
            tdm_load_rows((unsigned)(uintptr_t)ldsB +
                              (unsigned)(g * 16 * LDSB_STRIDE * 2),
                          Wt + (size_t)(g * HID + j0) * KTOT + IN_DIM, 16, KTOT);
        __builtin_amdgcn_s_wait_tensorcnt(0);
    }
#else
    stage_rows_manual(ldsA, hin, 64, HID);
#pragma unroll
    for (int g = 0; g < 4; ++g)
        stage_rows_manual(ldsB + (size_t)g * 16 * LDSB_STRIDE,
                          Wt + (size_t)(g * HID + j0) * KTOT + IN_DIM, 16, KTOT);
#endif
    __syncthreads();

    v8f acc[4] = {};
    v16bf a_cur = load_a_lds(ldsA, w * 16, 0);
    v16bf b0 = load_b_lds(ldsB,  0, 0);
    v16bf b1 = load_b_lds(ldsB, 16, 0);
    v16bf b2 = load_b_lds(ldsB, 32, 0);
    v16bf b3 = load_b_lds(ldsB, 48, 0);
    int k0 = 0;
    for (; k0 < HID - 32; k0 += 32) {
        v16bf a_n = load_a_lds(ldsA, w * 16, k0 + 32);
        v16bf c0 = load_b_lds(ldsB,  0, k0 + 32);
        v16bf c1 = load_b_lds(ldsB, 16, k0 + 32);
        v16bf c2 = load_b_lds(ldsB, 32, k0 + 32);
        v16bf c3 = load_b_lds(ldsB, 48, k0 + 32);
        acc[0] = wmma_bf16(a_cur, b0, acc[0]);
        acc[1] = wmma_bf16(a_cur, b1, acc[1]);
        acc[2] = wmma_bf16(a_cur, b2, acc[2]);
        acc[3] = wmma_bf16(a_cur, b3, acc[3]);
        a_cur = a_n; b0 = c0; b1 = c1; b2 = c2; b3 = c3;
    }
    acc[0] = wmma_bf16(a_cur, b0, acc[0]);
    acc[1] = wmma_bf16(a_cur, b1, acc[1]);
    acc[2] = wmma_bf16(a_cur, b2, acc[2]);
    acc[3] = wmma_bf16(a_cur, b3, acc[3]);

    int n  = lane & 15;
    int mb = (lane >> 4) << 3;
    int jn = j0 + n;
#pragma unroll
    for (int r = 0; r < 8; ++r) {
        int b = w * 16 + mb + r;
        size_t prow = (size_t)b * G4;
        float fg = acc[0][r] + (float)Pt[prow + 0 * HID + jn] + bias[0 * HID + jn];
        float ig = acc[1][r] + (float)Pt[prow + 1 * HID + jn] + bias[1 * HID + jn];
        float og = acc[2][r] + (float)Pt[prow + 2 * HID + jn] + bias[2 * HID + jn];
        float cg = acc[3][r] + (float)Pt[prow + 3 * HID + jn] + bias[3 * HID + jn];
        fg = sigmoidf_(fg);
        ig = sigmoidf_(ig);
        og = sigmoidf_(og);

        size_t hidx = (size_t)b * HID + jn;
        float cnew = fg * cbuf[hidx] + ig * tanhf(cg);
        float hnew = og * tanhf(cnew);

        out_t[hidx] = hfp[hidx];      // output[t] = h BEFORE update (fp32)
        hfp[hidx]   = hnew;
        cbuf[hidx]  = cnew;
        hout[hidx]  = (__bf16)hnew;   // feeds next step's WMMA
    }
}

// ---- Host side -------------------------------------------------------------

extern "C" void kernel_launch(void* const* d_in, const int* in_sizes, int n_in,
                              void* d_out, int out_size, void* d_ws, size_t ws_size,
                              hipStream_t stream) {
    (void)in_sizes; (void)n_in; (void)out_size; (void)ws_size;
    const float* x    = (const float*)d_in[0];
    const float* W    = (const float*)d_in[1];
    const float* bias = (const float*)d_in[2];
    const float* h0   = (const float*)d_in[3];
    const float* c0   = (const float*)d_in[4];
    float* out = (float*)d_out;

    const size_t WT_BYTES = (size_t)G4 * KTOT * sizeof(__bf16);                // 16 MB
    const size_t XB_BYTES = (size_t)T_STEPS * BATCH * IN_DIM * sizeof(__bf16); // 64 MB
    const size_t P_BYTES  = (size_t)T_STEPS * BATCH * G4 * sizeof(__bf16);     // 256 MB
    const size_t HB_BYTES = (size_t)BATCH * HID * sizeof(__bf16);              // 128 KB
    const size_t CF_BYTES = (size_t)BATCH * HID * sizeof(float);               // 256 KB

    char* ws = (char*)d_ws;
    __bf16* Wt  = (__bf16*)ws;                     ws += WT_BYTES;
    __bf16* xb  = (__bf16*)ws;                     ws += XB_BYTES;
    __bf16* P   = (__bf16*)ws;                     ws += P_BYTES;
    __bf16* hb0 = (__bf16*)ws;                     ws += HB_BYTES;
    __bf16* hb1 = (__bf16*)ws;                     ws += HB_BYTES;
    float*  hfp = (float*)ws;                      ws += CF_BYTES;
    float*  cbf = (float*)ws;                      ws += CF_BYTES;

    k_wt_convert<<<(KTOT * G4) / 256, 256, 0, stream>>>(W, Wt);
    k_x_convert<<<((size_t)T_STEPS * BATCH * IN_DIM / 4) / 256, 256, 0, stream>>>(x, xb);
    k_init<<<(BATCH * HID) / 256, 256, 0, stream>>>(h0, c0, hb0, hfp, cbf);

    k_xproj<<<64 * 256, 256, 0, stream>>>(xb, Wt, P);

    for (int t = 0; t < T_STEPS; ++t) {
        __bf16* hin  = (t & 1) ? hb1 : hb0;
        __bf16* hout = (t & 1) ? hb0 : hb1;
        k_step<<<HID / 16, 128, 0, stream>>>(
            hin, hout, hfp, cbf,
            P + (size_t)t * BATCH * G4, Wt, bias,
            out + (size_t)t * BATCH * HID);
    }
}
